// GNN_9088150798869
// MI455X (gfx1250) — matrix operand order repaired and checked by reading, not compiled, not run on hardware
//
#include <hip/hip_runtime.h>
#include <hip/hip_bf16.h>

typedef __attribute__((ext_vector_type(16))) _Float16 v16h;
typedef __attribute__((ext_vector_type(8)))  float    v8f;

#define D 64
#define NEG_SLOPE 0.2f

// ---------- helpers ----------
__device__ __forceinline__ unsigned encf(float f) {
    unsigned u = __float_as_uint(f);
    return (u & 0x80000000u) ? ~u : (u | 0x80000000u);   // monotone float->uint
}
__device__ __forceinline__ float decf(unsigned u) {
    unsigned b = (u & 0x80000000u) ? (u & 0x7FFFFFFFu) : ~u;
    return __uint_as_float(b);
}

// ---------- W (f32 64x64) -> f16, pre-swizzled into per-lane WMMA B fragments ----------
// Fragment (ks,t) for lane l occupies Wpk[((ks*4+t)*32 + l)*16 .. +15], so the GEMM
// loads each B fragment as one contiguous 32-byte v16h (2x global_load_b128).
// B layout (16-bit 32x16): lane -> N = l&15; elem i -> K = kbB+i (i<8), kbB+8+i (i>=8), kbB = ks*32+8*half.
__global__ void gat_cvt_w_pack(const float* __restrict__ W, _Float16* __restrict__ Wpk) {
    int idx = threadIdx.x;            // 256 threads = 8 fragments x 32 lanes
    int frag = idx >> 5;              // ks*4 + t
    int l    = idx & 31;
    int ks = frag >> 2, t = frag & 3;
    int hs = l >> 4, ln = l & 15;
    int kbB = ks * 32 + hs * 8;
    _Float16* dst = Wpk + (size_t)idx * 16;
    #pragma unroll
    for (int i = 0; i < 16; ++i)
        dst[i] = (_Float16)W[(kbB + i + (i >= 8 ? 8 : 0)) * D + t * 16 + ln];
}

// ---------- cE = dot(We, ae) ----------
__global__ void gat_cdot(const float* __restrict__ We, const float* __restrict__ ae,
                         float* __restrict__ cE) {
    __shared__ float sm[D];
    int t = threadIdx.x;
    sm[t] = We[t] * ae[t];
    __syncthreads();
    for (int o = D / 2; o > 0; o >>= 1) {
        if (t < o) sm[t] += sm[t + o];
        __syncthreads();
    }
    if (t == 0) cE[0] = sm[0];
}

// ---------- H = X @ W  (f16 WMMA, f32 accumulate). One wave per 16-row block. ----------
__global__ void gat_gemm_wmma(const float* __restrict__ X, const _Float16* __restrict__ Wpk,
                              float* __restrict__ H, int nblk, int nrows) {
    int wave = blockIdx.x * (blockDim.x >> 5) + (threadIdx.x >> 5);
    if (wave >= nblk) return;                 // uniform per wave: EXEC stays all-ones
    int l  = threadIdx.x & 31;
    int hs = l >> 4;                          // lane-half select (0 = lanes 0-15, 1 = 16-31)
    int ln = l & 15;
    int row = wave * 16 + ln;
    int rowL = row < nrows ? row : nrows - 1; // clamp (keeps EXEC full)

    const v16h*   Bfrag = (const v16h*)Wpk;
    const float4* xr4   = (const float4*)(X + (size_t)rowL * D);

    v8f acc0 = {}, acc1 = {}, acc2 = {}, acc3 = {};

    #pragma unroll
    for (int ks = 0; ks < 2; ++ks) {
        // A fragment: 16x32 f16. Lane needs K = {kbA..kbA+7} U {kbA+16..kbA+23}:
        // two contiguous 8-float runs -> 4x float4 loads, then pack to f16.
        int q = ks * 8 + hs * 2;              // float4 index of kbA within the row
        float4 p0 = xr4[q + 0];
        float4 p1 = xr4[q + 1];
        float4 p2 = xr4[q + 4];
        float4 p3 = xr4[q + 5];
        v16h a;
        a[0]  = (_Float16)p0.x; a[1]  = (_Float16)p0.y;
        a[2]  = (_Float16)p0.z; a[3]  = (_Float16)p0.w;
        a[4]  = (_Float16)p1.x; a[5]  = (_Float16)p1.y;
        a[6]  = (_Float16)p1.z; a[7]  = (_Float16)p1.w;
        a[8]  = (_Float16)p2.x; a[9]  = (_Float16)p2.y;
        a[10] = (_Float16)p2.z; a[11] = (_Float16)p2.w;
        a[12] = (_Float16)p3.x; a[13] = (_Float16)p3.y;
        a[14] = (_Float16)p3.z; a[15] = (_Float16)p3.w;

        #pragma unroll
        for (int t = 0; t < 4; ++t) {
            v16h b = Bfrag[(ks * 4 + t) * 32 + l];   // one 32B load (2x b128)
            v8f& acc = (t == 0) ? acc0 : (t == 1) ? acc1 : (t == 2) ? acc2 : acc3;
            acc = __builtin_amdgcn_wmma_f32_16x16x32_f16(
                false, a, false, b, (short)0, acc, false, false);
        }
    }
    // store D: VGPR r, lane -> M = r + 8*hs, N = 16*t + ln
    #pragma unroll
    for (int t = 0; t < 4; ++t) {
        const v8f& acc = (t == 0) ? acc0 : (t == 1) ? acc1 : (t == 2) ? acc2 : acc3;
        #pragma unroll
        for (int r = 0; r < 8; ++r) {
            int m = wave * 16 + r + hs * 8;
            if (m < nrows) H[(size_t)m * D + t * 16 + ln] = acc[r];
        }
    }
}

// ---------- per-node logits: S[n]=dot(h[n],a_src), Dv[n]=dot(h[n],a_dst) ----------
__global__ void gat_node_dots(const float* __restrict__ H, const float* __restrict__ as_,
                              const float* __restrict__ ad_, float* __restrict__ S,
                              float* __restrict__ Dv, int n) {
    int node = blockIdx.x * 8 + (threadIdx.x >> 5);
    int l = threadIdx.x & 31;
    if (node >= n) return;
    const float2 hv = ((const float2*)(H + (size_t)node * D))[l];
    const float2 av = ((const float2*)as_)[l];
    const float2 dv = ((const float2*)ad_)[l];
    float ps = hv.x * av.x + hv.y * av.y;
    float pd = hv.x * dv.x + hv.y * dv.y;
    #pragma unroll
    for (int o = 16; o > 0; o >>= 1) {
        ps += __shfl_xor(ps, o, 32);
        pd += __shfl_xor(pd, o, 32);
    }
    if (l == 0) { S[node] = ps; Dv[node] = pd; }
}

// ---------- pass 1: leaky-relu logits + segment max (atomicMax on monotone uint) ----------
__global__ void gat_edge_alpha(const long long* __restrict__ ei, const float* __restrict__ ea,
                               const float* __restrict__ S, const float* __restrict__ Dv,
                               const float* __restrict__ cE, float* __restrict__ abuf,
                               unsigned* __restrict__ menc, int E_) {
    int e = blockIdx.x * blockDim.x + threadIdx.x;
    if (e >= E_) return;
    int s = (int)ei[e], d = (int)ei[E_ + e];
    float a = S[s] + Dv[d] + ea[e] * cE[0];
    a = a > 0.f ? a : NEG_SLOPE * a;
    abuf[e] = a;
    atomicMax(menc + d, encf(a));
}

// ---------- pass 2: exp(alpha - m) + segment sum ----------
__global__ void gat_edge_exp(const long long* __restrict__ ei, float* __restrict__ buf,
                             const unsigned* __restrict__ menc, float* __restrict__ denom,
                             int E_) {
    int e = blockIdx.x * blockDim.x + threadIdx.x;
    if (e >= E_) return;
    int d = (int)ei[E_ + e];
    unsigned mu = menc[d];
    float m = (mu == 0u) ? 0.f : decf(mu);   // memset-0 init == "no edge"; ref maps -inf -> 0
    float w = __expf(buf[e] - m);
    buf[e] = w;
    atomicAdd(denom + d, w);
}

// ---------- pass 3: out[dst] += h[src] * w  (16 lanes per edge, float4 each) ----------
__global__ void gat_edge_scatter(const long long* __restrict__ ei, const float* __restrict__ buf,
                                 const float* __restrict__ denom, const float* __restrict__ H,
                                 float* __restrict__ out, int E_) {
    int gid = blockIdx.x * blockDim.x + threadIdx.x;
    int e = gid >> 4, p = gid & 15;
    if (e >= E_) return;
    int s = (int)ei[e], d = (int)ei[E_ + e];
    float w = buf[e] / (denom[d] + 1e-16f);
    float4 hv = ((const float4*)(H + (size_t)s * D))[p];
    float* o = out + (size_t)d * D + p * 4;
    atomicAdd(o + 0, hv.x * w);
    atomicAdd(o + 1, hv.y * w);
    atomicAdd(o + 2, hv.z * w);
    atomicAdd(o + 3, hv.w * w);
}

// ---------- epilogues ----------
__global__ void gat_bias_relu(float* __restrict__ a, const float* __restrict__ b, int n) {
    int i = blockIdx.x * blockDim.x + threadIdx.x;
    if (i < n) { float v = a[i] + b[i & (D - 1)]; a[i] = v > 0.f ? v : 0.f; }
}
__global__ void gat_bias_add(float* __restrict__ a, const float* __restrict__ b, int n) {
    int i = blockIdx.x * blockDim.x + threadIdx.x;
    if (i < n) a[i] += b[i & (D - 1)];
}

// ---------- host-side per-layer driver ----------
static void run_layer(const float* xin, const long long* ei, const float* ea,
                      const float* W, const float* We, const float* as_, const float* ad_,
                      const float* ae, float* outbuf,
                      float* h, float* sA, float* sD, unsigned* menc, float* denom,
                      float* ebuf, _Float16* Wpk, float* cE,
                      int N_, int E_, hipStream_t stream) {
    int nblk = (N_ + 15) / 16;
    gat_cvt_w_pack<<<1, 256, 0, stream>>>(W, Wpk);
    gat_cdot<<<1, D, 0, stream>>>(We, ae, cE);
    gat_gemm_wmma<<<(nblk + 3) / 4, 128, 0, stream>>>(xin, Wpk, h, nblk, N_);
    gat_node_dots<<<(N_ + 7) / 8, 256, 0, stream>>>(h, as_, ad_, sA, sD, N_);
    hipMemsetAsync(menc, 0, (size_t)N_ * 4, stream);
    hipMemsetAsync(denom, 0, (size_t)N_ * 4, stream);
    hipMemsetAsync(outbuf, 0, (size_t)N_ * D * 4, stream);
    gat_edge_alpha<<<(E_ + 255) / 256, 256, 0, stream>>>(ei, ea, sA, sD, cE, ebuf, menc, E_);
    gat_edge_exp<<<(E_ + 255) / 256, 256, 0, stream>>>(ei, ebuf, menc, denom, E_);
    long long tot = (long long)E_ * 16;
    gat_edge_scatter<<<(int)((tot + 255) / 256), 256, 0, stream>>>(ei, ebuf, denom, h, outbuf, E_);
}

extern "C" void kernel_launch(void* const* d_in, const int* in_sizes, int n_in,
                              void* d_out, int out_size, void* d_ws, size_t ws_size,
                              hipStream_t stream) {
    const float*     x   = (const float*)d_in[0];
    const long long* ei  = (const long long*)d_in[1];
    const float*     ea  = (const float*)d_in[2];
    const float* W1  = (const float*)d_in[3];
    const float* We1 = (const float*)d_in[4];
    const float* as1 = (const float*)d_in[5];
    const float* ad1 = (const float*)d_in[6];
    const float* ae1 = (const float*)d_in[7];
    const float* b1  = (const float*)d_in[8];
    const float* W2  = (const float*)d_in[9];
    const float* We2 = (const float*)d_in[10];
    const float* as2 = (const float*)d_in[11];
    const float* ad2 = (const float*)d_in[12];
    const float* ae2 = (const float*)d_in[13];
    const float* b2  = (const float*)d_in[14];
    float* out = (float*)d_out;

    const int N_ = in_sizes[0] / D;    // 100000
    const int E_ = in_sizes[2];        // 1600000

    // workspace carve-up (each section is a multiple of 32 bytes -> Wpk is 32B aligned)
    char* w = (char*)d_ws;
    float* h    = (float*)w;   w += (size_t)N_ * D * 4;   // projected features (reused per layer)
    float* agg  = (float*)w;   w += (size_t)N_ * D * 4;   // layer-1 aggregate -> relu in place
    float* sA   = (float*)w;   w += (size_t)N_ * 4;
    float* sD   = (float*)w;   w += (size_t)N_ * 4;
    unsigned* menc = (unsigned*)w; w += (size_t)N_ * 4;
    float* denom = (float*)w;  w += (size_t)N_ * 4;
    float* ebuf = (float*)w;   w += (size_t)E_ * 4;
    _Float16* Wpk = (_Float16*)w; w += (size_t)D * D * 2; // pre-swizzled f16 W fragments
    float* cE = (float*)w;     w += 32;

    // ----- layer 1 -----
    run_layer(x, ei, ea, W1, We1, as1, ad1, ae1, agg,
              h, sA, sD, menc, denom, ebuf, Wpk, cE, N_, E_, stream);
    gat_bias_relu<<<(N_ * D + 255) / 256, 256, 0, stream>>>(agg, b1, N_ * D);

    // ----- layer 2 (input = agg, output accumulated directly into d_out) -----
    run_layer(agg, ei, ea, W2, We2, as2, ad2, ae2, out,
              h, sA, sD, menc, denom, ebuf, Wpk, cE, N_, E_, stream);
    gat_bias_add<<<(N_ * D + 255) / 256, 256, 0, stream>>>(out, b2, N_ * D);
}
